// DownsampleQRNNAdapter_85581518340431
// MI455X (gfx1250) — compile-verified, matrix operand order checked by reading, not run on hardware
//
#include <hip/hip_runtime.h>
#include <hip/hip_bf16.h>

typedef _Float16 v16h __attribute__((ext_vector_type(16)));
typedef _Float16 v8h  __attribute__((ext_vector_type(8)));
typedef _Float16 v4h  __attribute__((ext_vector_type(4)));
typedef float    v8f  __attribute__((ext_vector_type(8)));

constexpr int D_   = 768;
constexpr int DS_  = 192;
constexpr int HW_  = 196;
constexpr int B_   = 4;
constexpr int S_   = 64;
constexpr int NROW = B_ * HW_;      // 784 (divisible by 16)
constexpr int M_   = S_ * NROW;     // 50176
constexpr int PER_W = DS_ * D_;     // 147456 (== 384*384 == 768*192)

__device__ __forceinline__ v8f zero8f() {
  v8f r;
#pragma unroll
  for (int i = 0; i < 8; ++i) r[i] = 0.0f;
  return r;
}

__device__ __forceinline__ v16h zero16h() {
  v16h r;
#pragma unroll
  for (int i = 0; i < 16; ++i) r[i] = (_Float16)0.0f;
  return r;
}

// Load one 16x32 f16 WMMA operand fragment for this lane.
// ISA 7.12.2 (16-bit A-matrix 16x32): lanes 0-15 hold K=[0..7]+[16..23],
// lanes 16-31 hold K=[8..15]+[24..31] of their row. `p` points at
// row_start + kb (kb multiple of 32); `h` = lane>>4.
__device__ __forceinline__ v16h load_frag(const _Float16* __restrict__ p, int h) {
  v8h lo = *reinterpret_cast<const v8h*>(p + h * 8);
  v8h hi = *reinterpret_cast<const v8h*>(p + 16 + h * 8);
  v16h r;
#pragma unroll
  for (int i = 0; i < 8; ++i) { r[i] = lo[i]; r[i + 8] = hi[i]; }
  return r;
}

#define WMMA_F16(a, b, c) \
  __builtin_amdgcn_wmma_f32_16x16x32_f16(false, (a), false, (b), (short)0, (c), false, false)

// ---------------------------------------------------------------------------
// Kernel 0: convert the three weight matrices fp32 -> fp16 (all 147456 elems)
// ---------------------------------------------------------------------------
__global__ void __launch_bounds__(256)
cvt_weights(const float* __restrict__ wd, const float* __restrict__ wq,
            const float* __restrict__ wu, _Float16* __restrict__ o) {
  int t = blockIdx.x * blockDim.x + threadIdx.x;
  if (t < PER_W)            o[t] = (_Float16)wd[t];
  else if (t < 2 * PER_W)   o[t] = (_Float16)wq[t - PER_W];
  else if (t < 3 * PER_W)   o[t] = (_Float16)wu[t - 2 * PER_W];
}

// ---------------------------------------------------------------------------
// Kernel 1: gather-transpose x[hw, b*S+s, d] -> A16[(s*784+n), d] fp16
// ---------------------------------------------------------------------------
__global__ void __launch_bounds__(256)
pack_x(const float* __restrict__ x, _Float16* __restrict__ A) {
  size_t tid = (size_t)blockIdx.x * blockDim.x + threadIdx.x;   // over M_*192 float4s
  if (tid >= (size_t)M_ * (D_ / 4)) return;
  int d4 = (int)(tid % (D_ / 4));
  size_t m = tid / (D_ / 4);
  int s  = (int)(m / NROW);
  int n  = (int)(m % NROW);
  int b  = n / HW_;
  int hw = n % HW_;
  const float4 v = *reinterpret_cast<const float4*>(
      x + ((size_t)hw * (B_ * S_) + (size_t)b * S_ + s) * D_ + (size_t)d4 * 4);
  v4h o = { (_Float16)v.x, (_Float16)v.y, (_Float16)v.z, (_Float16)v.w };
  *reinterpret_cast<v4h*>(A + m * D_ + (size_t)d4 * 4) = o;
}

// ---------------------------------------------------------------------------
// Kernel 2: GEMM1  XD16[M,192] = A16[M,768] @ Wd16^T + bd      (fp16 out)
// Each wave: 16x64 output strip (4 WMMA tiles), K-loop 768/32 = 24 iters.
// ---------------------------------------------------------------------------
__global__ void __launch_bounds__(256)
gemm_down(const _Float16* __restrict__ A, const _Float16* __restrict__ W,
          const float* __restrict__ bias, _Float16* __restrict__ C) {
  const int K = D_, N = DS_;
  int wave = (blockIdx.x * blockDim.x + threadIdx.x) >> 5;
  int lane = threadIdx.x & 31;
  const int nw = N / 64;                         // 3 wave-columns
  int mt = wave / nw, ntw = wave % nw;
  if (mt >= M_ / 16) return;
  int m0 = mt * 16, n0 = ntw * 64;
  int half = lane >> 4, idx = lane & 15;

  v8f acc[4];
#pragma unroll
  for (int t = 0; t < 4; ++t) acc[t] = zero8f();

  const _Float16* arow = A + (size_t)(m0 + idx) * K;
  for (int kb = 0; kb < K; kb += 32) {
    v16h a = load_frag(arow + kb, half);
#pragma unroll
    for (int t = 0; t < 4; ++t) {
      const _Float16* bcol = W + (size_t)(n0 + t * 16 + idx) * K + kb;
      v16h b = load_frag(bcol, half);
      acc[t] = WMMA_F16(a, b, acc[t]);
    }
  }
#pragma unroll
  for (int t = 0; t < 4; ++t) {
    int n = n0 + t * 16 + idx;
    float bs = bias[n];
#pragma unroll
    for (int r = 0; r < 8; ++r) {
      int m = m0 + half * 8 + r;
      C[(size_t)m * N + n] = (_Float16)(acc[t][r] + bs);
    }
  }
}

// ---------------------------------------------------------------------------
// Kernel 3: GEMM2 (QRNN gate GEMM, lookback folded into addressing)
//   zf[m, :] = [xd[s-1,n,:], xd[s,n,:]] @ Wq^T + bq
// Each wave computes the matched (z, f) 16x16 column pair, then fuses
//   a = sigmoid(f); zt = (1-a)*tanh(z)   -> AF16, ZT16 (fp16)
// ---------------------------------------------------------------------------
__global__ void __launch_bounds__(256)
gemm_qrnn(const _Float16* __restrict__ XD, const _Float16* __restrict__ Wq,
          const float* __restrict__ bias, _Float16* __restrict__ AF,
          _Float16* __restrict__ ZT) {
  const int K = 2 * DS_;                         // 384
  int wave = (blockIdx.x * blockDim.x + threadIdx.x) >> 5;
  int lane = threadIdx.x & 31;
  const int nw = DS_ / 16;                       // 12 z-tiles
  int mt = wave / nw, nt = wave % nw;
  if (mt >= M_ / 16) return;
  int m0 = mt * 16;
  int s  = m0 / NROW;                            // whole tile shares s (784 % 16 == 0)
  int nb = m0 % NROW;
  int half = lane >> 4, idx = lane & 15;

  v8f accZ = zero8f(), accF = zero8f();

  const _Float16* cur  = XD + ((size_t)s * NROW + nb + idx) * DS_;
  const _Float16* prev = cur - (size_t)NROW * DS_;   // valid only if s > 0
  int nzc = nt * 16 + idx;                           // z column
  const _Float16* bz = Wq + (size_t)nzc * K;
  const _Float16* bf = Wq + (size_t)(nzc + DS_) * K;

  for (int kb = 0; kb < K; kb += 32) {
    v16h a;
    if (kb < DS_) {
      a = (s == 0) ? zero16h() : load_frag(prev + kb, half);
    } else {
      a = load_frag(cur + (kb - DS_), half);
    }
    v16h b0 = load_frag(bz + kb, half);
    v16h b1 = load_frag(bf + kb, half);
    accZ = WMMA_F16(a, b0, accZ);
    accF = WMMA_F16(a, b1, accF);
  }

  float bsz = bias[nzc], bsf = bias[nzc + DS_];
#pragma unroll
  for (int r = 0; r < 8; ++r) {
    int m = m0 + half * 8 + r;
    float z = accZ[r] + bsz;
    float f = accF[r] + bsf;
    float a  = 1.0f / (1.0f + __expf(-f));
    float zt = (1.0f - a) * tanhf(z);
    size_t o = (size_t)m * DS_ + nzc;
    AF[o] = (_Float16)a;
    ZT[o] = (_Float16)zt;
  }
}

// ---------------------------------------------------------------------------
// Kernel 4: fo_pool scan  h_t = a_t * h_{t-1} + zt_t  (sequential over S=64,
// parallel over 784*192 = 150528 coalesced lanes). Writes H16 fp16.
// ---------------------------------------------------------------------------
__global__ void __launch_bounds__(256)
fo_scan(const _Float16* __restrict__ AF, const _Float16* __restrict__ ZT,
        _Float16* __restrict__ H) {
  int tid = blockIdx.x * blockDim.x + threadIdx.x;
  if (tid >= NROW * DS_) return;
  float h = 0.0f;
#pragma unroll 4
  for (int s = 0; s < S_; ++s) {
    size_t i = (size_t)s * (NROW * DS_) + tid;
    float a  = (float)AF[i];
    float zt = (float)ZT[i];
    h = a * h + zt;
    H[i] = (_Float16)h;
  }
}

// ---------------------------------------------------------------------------
// Kernel 5: GEMM3  out = H16[M,192] @ Wu16^T + bu + skip(x), with the inverse
// reshape (s, b*hw, d) -> (hw, b*S+s, d) fused into the fp32 store.
// ---------------------------------------------------------------------------
__global__ void __launch_bounds__(256)
gemm_up(const _Float16* __restrict__ H, const _Float16* __restrict__ W,
        const float* __restrict__ bias, const float* __restrict__ x,
        float* __restrict__ out) {
  const int K = DS_, N = D_;
  int wave = (blockIdx.x * blockDim.x + threadIdx.x) >> 5;
  int lane = threadIdx.x & 31;
  const int nw = N / 64;                         // 12 wave-columns
  int mt = wave / nw, ntw = wave % nw;
  if (mt >= M_ / 16) return;
  int m0 = mt * 16, n0 = ntw * 64;
  int s  = m0 / NROW;
  int nb = m0 % NROW;
  int half = lane >> 4, idx = lane & 15;

  v8f acc[4];
#pragma unroll
  for (int t = 0; t < 4; ++t) acc[t] = zero8f();

  const _Float16* arow = H + (size_t)(m0 + idx) * K;
  for (int kb = 0; kb < K; kb += 32) {
    v16h a = load_frag(arow + kb, half);
#pragma unroll
    for (int t = 0; t < 4; ++t) {
      const _Float16* bcol = W + (size_t)(n0 + t * 16 + idx) * K + kb;
      v16h b = load_frag(bcol, half);
      acc[t] = WMMA_F16(a, b, acc[t]);
    }
  }
#pragma unroll
  for (int t = 0; t < 4; ++t) {
    int n = n0 + t * 16 + idx;
    float bs = bias[n];
#pragma unroll
    for (int r = 0; r < 8; ++r) {
      int nrow = nb + half * 8 + r;
      int b  = nrow / HW_;
      int hw = nrow % HW_;
      size_t o = ((size_t)hw * (B_ * S_) + (size_t)b * S_ + s) * D_ + n;
      out[o] = acc[t][r] + bs + x[o];
    }
  }
}

// ---------------------------------------------------------------------------
extern "C" void kernel_launch(void* const* d_in, const int* in_sizes, int n_in,
                              void* d_out, int out_size, void* d_ws, size_t ws_size,
                              hipStream_t stream) {
  const float* x  = (const float*)d_in[0];
  const float* Wd = (const float*)d_in[1];
  const float* bd = (const float*)d_in[2];
  const float* Wq = (const float*)d_in[3];
  const float* bq = (const float*)d_in[4];
  const float* Wu = (const float*)d_in[5];
  const float* bu = (const float*)d_in[6];
  float* out = (float*)d_out;

  // Workspace layout (A16 region recycled for AF/ZT/H after GEMM1):
  char* ws = (char*)d_ws;
  _Float16* W16  = (_Float16*)ws;                        // 3*147456 f16
  _Float16* Wd16 = W16;
  _Float16* Wq16 = W16 + PER_W;
  _Float16* Wu16 = W16 + 2 * PER_W;
  size_t off = ((size_t)3 * PER_W * sizeof(_Float16) + 255) & ~(size_t)255;
  _Float16* A16  = (_Float16*)(ws + off);                // M_*768 f16 (77 MB)
  off += (size_t)M_ * D_ * sizeof(_Float16);
  _Float16* XD16 = (_Float16*)(ws + off);                // M_*192 f16 (19.3 MB)
  // after GEMM1, the A16 region is dead -> reuse for AF/ZT/H (3 * 19.3 MB)
  _Float16* AF16 = A16;
  _Float16* ZT16 = A16 + (size_t)M_ * DS_;
  _Float16* H16  = A16 + (size_t)2 * M_ * DS_;

  const int TPB = 256;

  // 0) weights fp32 -> fp16
  cvt_weights<<<(3 * PER_W + TPB - 1) / TPB, TPB, 0, stream>>>(Wd, Wq, Wu, W16);

  // 1) gather-transpose + fp16 pack of x
  {
    size_t total = (size_t)M_ * (D_ / 4);
    pack_x<<<(unsigned)((total + TPB - 1) / TPB), TPB, 0, stream>>>(x, A16);
  }

  // 2) GEMM1: xd = xr @ Wd^T + bd
  {
    int waves = (M_ / 16) * (DS_ / 64);                  // 3136 * 3
    gemm_down<<<(waves * 32 + TPB - 1) / TPB, TPB, 0, stream>>>(A16, Wd16, bd, XD16);
  }

  // 3) GEMM2 + gate activations
  {
    int waves = (M_ / 16) * (DS_ / 16);                  // 3136 * 12
    gemm_qrnn<<<(waves * 32 + TPB - 1) / TPB, TPB, 0, stream>>>(XD16, Wq16, bq, AF16, ZT16);
  }

  // 4) fo_pool sequential scan over S
  fo_scan<<<(NROW * DS_ + TPB - 1) / TPB, TPB, 0, stream>>>(AF16, ZT16, H16);

  // 5) GEMM3: out = h @ Wu^T + bu + skip, with inverse reshape fused
  {
    int waves = (M_ / 16) * (D_ / 64);                   // 3136 * 12
    gemm_up<<<(waves * 32 + TPB - 1) / TPB, TPB, 0, stream>>>(H16, Wu16, bu, x, out);
  }
}
